// GnnLayer_26568667693807
// MI455X (gfx1250) — matrix operand order repaired and checked by reading, not compiled, not run on hardware
//
#include <hip/hip_runtime.h>
#include <hip/hip_bf16.h>

typedef _Float16 half_t;
typedef __attribute__((ext_vector_type(16))) _Float16 v16h;
typedef __attribute__((ext_vector_type(8)))  float    v8f;

#define B_   4
#define N_   32768
#define KNB  16
#define LDT  72     // padded LDS row stride (halfs) for the relayout tiles
#define NFRAG 20    // 8 (Wk1) + 8 (Wk2) + 4 (Wk3) B-fragments

__device__ __forceinline__ float gelu_exact(float x) {
  // exact GELU: 0.5*x*(1+erf(x/sqrt(2)))
  return 0.5f * x * (1.0f + erff(x * 0.70710678118654752440f));
}

__device__ __forceinline__ v8f wmma16(v16h a, v16h b, v8f c) {
  // D = A(16x32 f16) * B(32x16 f16) + C(16x16 f32)
  return __builtin_amdgcn_wmma_f32_16x16x32_f16(false, a, false, b, (short)0, c,
                                                false, false);
}

// B fragment (32Kx16N) from f32 row-major weight [krows x ld]; rows >= krows are 0.
// Layout: N=lane&15; elem j -> K = (j&7) + 16*(j>=8) + 8*(lane>=16).
__device__ __forceinline__ v16h load_bfrag(const float* __restrict__ W, int ld,
                                           int kofs, int nofs, int krows, int lane) {
  v16h b;
  const int n = nofs + (lane & 15);
  const int kbase = kofs + ((lane & 16) ? 8 : 0);
#pragma unroll
  for (int j = 0; j < 16; ++j) {
    const int k = kbase + (j & 7) + ((j >= 8) ? 16 : 0);
    b[j] = (k < krows) ? (half_t)W[k * ld + n] : (half_t)0.0f;
  }
  return b;
}

// A fragment (16Mx32K) from per-wave LDS tile [16 x LDT] (f16), K offset kofs.
// Layout: M = lane&15; elem pair v -> K = kofs + 2*(v&3) + 16*(v>=4) + 8*(lane>=16).
__device__ __forceinline__ v16h load_afrag(const half_t* tile, int kofs, int lane) {
  v16h a;
  const int m = lane & 15;
  const int khi = (lane & 16) ? 8 : 0;
  const half_t* row = tile + m * LDT;
#pragma unroll
  for (int v = 0; v < 8; ++v) {
    const int k = kofs + 2 * (v & 3) + ((v >= 4) ? 16 : 0) + khi;
    a[2 * v]     = row[k];
    a[2 * v + 1] = row[k + 1];
  }
  return a;
}

// Store a 16x16 f32 accumulator tile into the LDS relayout tile as f16,
// adding bias and (optionally) exact GELU. C/D layout: col = nofs + (lane&15),
// row = j + 8*(lane>=16).  Each (row,col) written by exactly one lane.
__device__ __forceinline__ void store_acc(half_t* tile, v8f c, float bias,
                                          int nofs, int lane, bool act) {
  const int col = nofs + (lane & 15);
  const int mbase = (lane & 16) ? 8 : 0;
#pragma unroll
  for (int j = 0; j < 8; ++j) {
    float v = c[j] + bias;
    if (act) v = gelu_exact(v);
    tile[(mbase + j) * LDT + col] = (half_t)v;
  }
}

// ---------------------------------------------------------------------------
// Kernel 1: x = gelu(inp @ Wp1 + bp1) @ Wp2 + bp2   for all (b,n)
// Writes x to d_out as f32 (residual source) and to d_ws as f16 (gather source)
// ---------------------------------------------------------------------------
__global__ __launch_bounds__(256) void pointwise_mlp(
    const float* __restrict__ inp, const float* __restrict__ Wp1,
    const float* __restrict__ bp1, const float* __restrict__ Wp2,
    const float* __restrict__ bp2, float* __restrict__ x32,
    half_t* __restrict__ x16) {
  __shared__ half_t lds[8][16 * LDT];
  const int lane = threadIdx.x & 31;
  half_t* tile = lds[threadIdx.x >> 5];
  const int wave = (int)((blockIdx.x * blockDim.x + threadIdx.x) >> 5);
  const int nwaves = (int)((gridDim.x * blockDim.x) >> 5);

  // Weights resident in VGPRs (only 8 fragments here: cheap).
  v16h w1[4], w2[2][2];
#pragma unroll
  for (int nt = 0; nt < 4; ++nt) w1[nt] = load_bfrag(Wp1, 64, 0, nt * 16, 32, lane);
#pragma unroll
  for (int kt = 0; kt < 2; ++kt)
#pragma unroll
    for (int nt = 0; nt < 2; ++nt)
      w2[kt][nt] = load_bfrag(Wp2, 32, kt * 32, nt * 16, 64, lane);
  float b1v[4], b2v[2];
#pragma unroll
  for (int nt = 0; nt < 4; ++nt) b1v[nt] = bp1[nt * 16 + (lane & 15)];
#pragma unroll
  for (int nt = 0; nt < 2; ++nt) b2v[nt] = bp2[nt * 16 + (lane & 15)];

  const int ntiles = (B_ * N_) / 16;
  for (int t = wave; t < ntiles; t += nwaves) {
    const int base = t * 16;
    // Prefetch next tile's row for this lane into L2 (global_prefetch_b8).
    if (t + nwaves < ntiles)
      __builtin_prefetch(inp + (size_t)((t + nwaves) * 16 + (lane & 15)) * 32, 0, 0);
    // A fragment directly from global (16 rows x 32 feats, f32 -> f16)
    v16h a;
    {
      const int m = lane & 15;
      const int khi = (lane & 16) ? 8 : 0;
      const float* rp = inp + (size_t)(base + m) * 32;
#pragma unroll
      for (int v = 0; v < 8; ++v) {
        const int k = 2 * (v & 3) + ((v >= 4) ? 16 : 0) + khi;
        a[2 * v]     = (half_t)rp[k];
        a[2 * v + 1] = (half_t)rp[k + 1];
      }
    }
    // layer 1: 16x32 @ 32x64 -> 16x64, gelu, relayout via LDS
#pragma unroll
    for (int nt = 0; nt < 4; ++nt) {
      v8f c = {};
      c = wmma16(a, w1[nt], c);
      store_acc(tile, c, b1v[nt], nt * 16, lane, true);
    }
    v16h a2lo = load_afrag(tile, 0, lane);   // LDS ops in-order per wave
    v16h a2hi = load_afrag(tile, 32, lane);
    // layer 2: 16x64 @ 64x32 -> 16x32, write f32 + f16
#pragma unroll
    for (int nt = 0; nt < 2; ++nt) {
      v8f c = {};
      c = wmma16(a2lo, w2[0][nt], c);
      c = wmma16(a2hi, w2[1][nt], c);
      const int col = nt * 16 + (lane & 15);
      const int mbase = (lane & 16) ? 8 : 0;
#pragma unroll
      for (int j = 0; j < 8; ++j) {
        const float v = c[j] + b2v[nt];
        const size_t o = (size_t)(base + mbase + j) * 32 + col;
        x32[o] = v;
        x16[o] = (half_t)v;
      }
    }
  }
}

// ---------------------------------------------------------------------------
// Kernel 2: per node (b,n): gather K=16 neighbors -> 16x36 agg tile (padded to
// 64), 36->64->64->32 MLP via WMMA, mean over the 16 rows, residual + LN.
// One WMMA M-tile == one node's neighbor set.  The 20 weight B-fragments are
// staged once per block into LDS as per-lane 32B images (2x ds_load_b128 per
// use) to keep VGPR pressure low and occupancy high.
// ---------------------------------------------------------------------------
__global__ __launch_bounds__(256) void neighbor_mlp(
    const half_t* __restrict__ x16, const float* __restrict__ igrid,
    const float* __restrict__ ogrid, const int* __restrict__ nbr,
    const float* __restrict__ Wk1, const float* __restrict__ bk1,
    const float* __restrict__ Wk2, const float* __restrict__ bk2,
    const float* __restrict__ Wk3, const float* __restrict__ bk3,
    const float* __restrict__ gamma, const float* __restrict__ beta,
    float* xout /* holds x on entry; overwritten with final output */) {
  __shared__ half_t lds[8][16 * LDT];               // per-wave relayout tiles
  __shared__ __align__(32) half_t wlds[NFRAG][32][16];  // weight fragment images
  const int lane = threadIdx.x & 31;
  half_t* tile = lds[threadIdx.x >> 5];
  const int wave = (int)((blockIdx.x * blockDim.x + threadIdx.x) >> 5);
  const int nwaves = (int)((gridDim.x * blockDim.x) >> 5);

  // Stage all 20 B-fragments into LDS (block-cooperative, once).
  // frag id: [0..7]=Wk1 (kt*4+nt), [8..15]=Wk2, [16..19]=Wk3 (kt*2+nt)
  for (int s = threadIdx.x; s < NFRAG * 32; s += blockDim.x) {
    const int f = s >> 5, l = s & 31;
    const float* W;
    int ld, kofs, nofs, krows;
    if (f < 8)       { W = Wk1; ld = 64; kofs = (f >> 2) * 32;        nofs = (f & 3) * 16;        krows = 36; }
    else if (f < 16) { W = Wk2; ld = 64; kofs = ((f - 8) >> 2) * 32;  nofs = ((f - 8) & 3) * 16;  krows = 64; }
    else             { W = Wk3; ld = 32; kofs = ((f - 16) >> 1) * 32; nofs = ((f - 16) & 1) * 16; krows = 64; }
    const int n = nofs + (l & 15);
    const int kbase = kofs + ((l & 16) ? 8 : 0);
#pragma unroll
    for (int j = 0; j < 16; ++j) {
      const int k = kbase + (j & 7) + ((j >= 8) ? 16 : 0);
      wlds[f][l][j] = (k < krows) ? (half_t)W[k * ld + n] : (half_t)0.0f;
    }
  }
  __syncthreads();

  float bAv[4], bBv[4], bCv[2];
#pragma unroll
  for (int nt = 0; nt < 4; ++nt) {
    bAv[nt] = bk1[nt * 16 + (lane & 15)];
    bBv[nt] = bk2[nt * 16 + (lane & 15)];
  }
#pragma unroll
  for (int nt = 0; nt < 2; ++nt) bCv[nt] = bk3[nt * 16 + (lane & 15)];

  const int col0 = lane & 15;
  const float g0 = gamma[col0], g1 = gamma[16 + col0];
  const float be0 = beta[col0], be1 = beta[16 + col0];

  const int total = B_ * N_;
  for (int gidx = wave; gidx < total; gidx += nwaves) {
    const int b = gidx >> 15;        // gidx / N_
    const int n = gidx & (N_ - 1);   // gidx % N_
    const int* idxp = nbr + (size_t)n * KNB;

    // Prefetch next node's index row + residual row into L2.
    const int nxt = gidx + nwaves;
    if (nxt < total) {
      __builtin_prefetch(nbr + (size_t)(nxt & (N_ - 1)) * KNB, 0, 0);
      __builtin_prefetch(xout + (size_t)nxt * 32, 0, 0);
    }

    // Stage agg row r = [y_pos(2), x_pos(2), f_y(32), zeros(28)] into LDS.
    const int r = lane & 15;
    const int nb = idxp[r];
    const half_t* fy = x16 + ((size_t)b * N_ + nb) * 32;
    half_t* row = tile + r * LDT;
    if (lane < 16) {
      row[0] = (half_t)igrid[nb * 2 + 0];
      row[1] = (half_t)igrid[nb * 2 + 1];
      row[2] = (half_t)ogrid[n * 2 + 0];
      row[3] = (half_t)ogrid[n * 2 + 1];
#pragma unroll
      for (int f = 0; f < 16; ++f) row[4 + f] = fy[f];
    } else {
#pragma unroll
      for (int f = 16; f < 32; ++f) row[4 + f] = fy[f];
#pragma unroll
      for (int f = 36; f < 64; ++f) row[f] = (half_t)0.0f;
    }

    // layer 1: 16x64(padded agg) @ 64x64
    v16h aLo = load_afrag(tile, 0, lane);
    v16h aHi = load_afrag(tile, 32, lane);
#pragma unroll
    for (int nt = 0; nt < 4; ++nt) {
      v8f c = {};
      c = wmma16(aLo, *(const v16h*)wlds[nt][lane], c);
      c = wmma16(aHi, *(const v16h*)wlds[4 + nt][lane], c);
      store_acc(tile, c, bAv[nt], nt * 16, lane, true);
    }
    // layer 2: 16x64 @ 64x64
    v16h hLo = load_afrag(tile, 0, lane);
    v16h hHi = load_afrag(tile, 32, lane);
#pragma unroll
    for (int nt = 0; nt < 4; ++nt) {
      v8f c = {};
      c = wmma16(hLo, *(const v16h*)wlds[8 + nt][lane], c);
      c = wmma16(hHi, *(const v16h*)wlds[12 + nt][lane], c);
      store_acc(tile, c, bBv[nt], nt * 16, lane, true);
    }
    // layer 3: 16x64 @ 64x32, then mean over the 16 rows (neighbors)
    v16h kLo = load_afrag(tile, 0, lane);
    v16h kHi = load_afrag(tile, 32, lane);
    float mean[2];
#pragma unroll
    for (int nt = 0; nt < 2; ++nt) {
      v8f c = {};
      c = wmma16(kLo, *(const v16h*)wlds[16 + nt][lane], c);
      c = wmma16(kHi, *(const v16h*)wlds[18 + nt][lane], c);
      float s = 0.0f;
#pragma unroll
      for (int j = 0; j < 8; ++j) s += c[j];
      s += __shfl_xor(s, 16, 32);      // other 8 rows live in lane^16
      mean[nt] = s * (1.0f / 16.0f) + bCv[nt];
    }

    // residual + layernorm over 32 features (lane l<16 owns feats l and l+16)
    const size_t obase = (size_t)gidx * 32;
    float v0 = mean[0] + xout[obase + col0];
    float v1 = mean[1] + xout[obase + 16 + col0];
    float s = v0 + v1;
#pragma unroll
    for (int o = 1; o < 16; o <<= 1) s += __shfl_xor(s, o, 32);
    const float mu = s * (1.0f / 32.0f);
    const float d0 = v0 - mu, d1 = v1 - mu;
    float q = d0 * d0 + d1 * d1;
#pragma unroll
    for (int o = 1; o < 16; o <<= 1) q += __shfl_xor(q, o, 32);
    const float rstd = rsqrtf(q * (1.0f / 32.0f) + 1e-5f);
    if (lane < 16) {
      xout[obase + col0]      = d0 * rstd * g0 + be0;
      xout[obase + 16 + col0] = d1 * rstd * g1 + be1;
    }
  }
}

extern "C" void kernel_launch(void* const* d_in, const int* in_sizes, int n_in,
                              void* d_out, int out_size, void* d_ws, size_t ws_size,
                              hipStream_t stream) {
  (void)in_sizes; (void)n_in; (void)out_size; (void)ws_size;
  const float* inp   = (const float*)d_in[0];
  const float* igrid = (const float*)d_in[1];
  const float* ogrid = (const float*)d_in[2];
  const int*   nbr   = (const int*)d_in[3];
  const float* Wp1 = (const float*)d_in[4];
  const float* bp1 = (const float*)d_in[5];
  const float* Wp2 = (const float*)d_in[6];
  const float* bp2 = (const float*)d_in[7];
  const float* Wk1 = (const float*)d_in[8];
  const float* bk1 = (const float*)d_in[9];
  const float* Wk2 = (const float*)d_in[10];
  const float* bk2 = (const float*)d_in[11];
  const float* Wk3 = (const float*)d_in[12];
  const float* bk3 = (const float*)d_in[13];
  const float* gam = (const float*)d_in[14];
  const float* bet = (const float*)d_in[15];
  float*  out = (float*)d_out;      // holds x after kernel1, final output after kernel2
  half_t* x16 = (half_t*)d_ws;      // f16 copy of x for the WMMA gather path (8 MB)

  pointwise_mlp<<<512, 256, 0, stream>>>(inp, Wp1, bp1, Wp2, bp2, out, x16);
  neighbor_mlp<<<1024, 256, 0, stream>>>(x16, igrid, ogrid, nbr,
                                         Wk1, bk1, Wk2, bk2, Wk3, bk3,
                                         gam, bet, out);
}
